// KerasMoHAttention_70334384439647
// MI455X (gfx1250) — matrix operand order repaired and checked by reading, not compiled, not run on hardware
//
#include <hip/hip_runtime.h>
#include <hip/hip_bf16.h>
#include <math.h>

// ---------------------------------------------------------------------------
// Problem constants (B=2, S=2048, D=1024, H=16, hd=64, SH=1, RH=3)
// ---------------------------------------------------------------------------
constexpr int B_  = 2;
constexpr int S_  = 2048;
constexpr int D_  = 1024;
constexpr int H_  = 16;
constexpr int HD_ = 64;
constexpr int M_  = B_ * S_;          // 4096 token rows

typedef __attribute__((ext_vector_type(16))) __bf16 v16bf;
typedef __attribute__((ext_vector_type(8)))  float  v8f;
typedef __attribute__((ext_vector_type(4)))  int    v4i;

__device__ __forceinline__ v8f wmma_bf16(v16bf a, v16bf b, v8f c) {
  return __builtin_amdgcn_wmma_f32_16x16x32_bf16(false, a, false, b, (short)0, c,
                                                 false, false);
}

union frag_u { v16bf v; uint4 q[2]; };

// A-fragment (16x32 bf16): lanes 0-15 hold K=0..7 & 16..23 of their row,
// lanes 16-31 hold K=8..15 & 24..31.  rowbase points at element [row][0].
__device__ __forceinline__ v16bf frag_a(const __bf16* rowbase, int lane) {
  const int klo = (lane >> 4) << 3;
  frag_u u;
  u.q[0] = *(const uint4*)(rowbase + klo);
  u.q[1] = *(const uint4*)(rowbase + klo + 16);
  return u.v;
}

// B-fragment (32x16 bf16) from transposed tile BT[n][k]: lanes 0-15 hold
// K=0..7 & 16..23 of their column, lanes 16-31 hold K=8..15 & 24..31.
__device__ __forceinline__ v16bf frag_b(const __bf16* colbase, int lane) {
  const int klo = (lane >> 4) << 3;
  frag_u u;
  u.q[0] = *(const uint4*)(colbase + klo);
  u.q[1] = *(const uint4*)(colbase + klo + 16);
  return u.v;
}

// ---------------------------------------------------------------------------
// 16-lane row reductions via DPP16 (no LDS round-trip).
// ---------------------------------------------------------------------------
#if __has_builtin(__builtin_amdgcn_update_dpp)
template <int CTRL>
__device__ __forceinline__ float dppmov(float x) {
  return __int_as_float(__builtin_amdgcn_update_dpp(
      0, __float_as_int(x), CTRL, 0xf, 0xf, true));
}
__device__ __forceinline__ float rmax16(float x) {
  x = fmaxf(x, dppmov<0xB1>(x));   // xor1: quad_perm(1,0,3,2)
  x = fmaxf(x, dppmov<0x4E>(x));   // xor2: quad_perm(2,3,0,1)
  x = fmaxf(x, dppmov<0x141>(x));  // row_half_mirror
  x = fmaxf(x, dppmov<0x140>(x));  // row_mirror
  return x;
}
__device__ __forceinline__ float rsum16(float x) {
  x += dppmov<0xB1>(x);
  x += dppmov<0x4E>(x);
  x += dppmov<0x141>(x);
  x += dppmov<0x140>(x);
  return x;
}
#else
__device__ __forceinline__ float rmax16(float x) {
#pragma unroll
  for (int m = 8; m >= 1; m >>= 1) x = fmaxf(x, __shfl_xor(x, m, 16));
  return x;
}
__device__ __forceinline__ float rsum16(float x) {
#pragma unroll
  for (int m = 8; m >= 1; m >>= 1) x += __shfl_xor(x, m, 16);
  return x;
}
#endif

// ---------------------------------------------------------------------------
// CDNA5 async HBM -> LDS copies (ASYNCcnt-tracked).
// ---------------------------------------------------------------------------
#define GAS __attribute__((address_space(1)))
#define LAS __attribute__((address_space(3)))

__device__ __forceinline__ void async_copy_b128(void* gsrc, void* ldst) {
#if __has_builtin(__builtin_amdgcn_global_load_async_to_lds_b128)
  __builtin_amdgcn_global_load_async_to_lds_b128((GAS v4i*)gsrc,
                                                 (LAS v4i*)ldst, 0, 0);
#else
  asm volatile("global_load_async_to_lds_b128 %0, %1, off"
               :: "v"((unsigned)(size_t)ldst),
                  "v"((unsigned long long)(size_t)gsrc)
               : "memory");
#endif
}
__device__ __forceinline__ void wait_async_le2() {
#if __has_builtin(__builtin_amdgcn_s_wait_asynccnt)
  __builtin_amdgcn_s_wait_asynccnt(2);
#else
  asm volatile("s_wait_asynccnt 0x2" ::: "memory");
#endif
}
__device__ __forceinline__ void wait_async_le4() {
#if __has_builtin(__builtin_amdgcn_s_wait_asynccnt)
  __builtin_amdgcn_s_wait_asynccnt(4);
#else
  asm volatile("s_wait_asynccnt 0x4" ::: "memory");
#endif
}

// ---------------------------------------------------------------------------
// One-time format conversions.
// ---------------------------------------------------------------------------
__global__ __launch_bounds__(256, 4)
void cvt_f32_to_bf16(const float* __restrict__ s, __bf16* __restrict__ d) {
  const size_t i = (size_t)blockIdx.x * blockDim.x + threadIdx.x;
  float4 f = *(const float4*)(s + i * 4);
  __bf16 t[4] = {(__bf16)f.x, (__bf16)f.y, (__bf16)f.z, (__bf16)f.w};
  *(uint2*)(d + i * 4) = *(const uint2*)t;
}

// src fp32 [D][D] row-major -> dst bf16 [D][D] transposed (dst[n][k]=src[k][n])
__global__ __launch_bounds__(256, 4)
void transpose_to_bf16(const float* __restrict__ src, __bf16* __restrict__ dst) {
  __shared__ float tile[32][33];
  const int tx = threadIdx.x & 31, ty = threadIdx.x >> 5;  // 0..7
  const int c0 = blockIdx.x * 32, r0 = blockIdx.y * 32;
#pragma unroll
  for (int i = 0; i < 32; i += 8)
    tile[ty + i][tx] = src[(size_t)(r0 + ty + i) * D_ + c0 + tx];
  __syncthreads();
#pragma unroll
  for (int i = 0; i < 32; i += 8)
    dst[(size_t)(c0 + ty + i) * D_ + r0 + tx] = (__bf16)tile[tx][ty + i];
}

// ---------------------------------------------------------------------------
// All-bf16 GEMM with async double-buffered staging:
//   C = diag(rowscale)? * A[M,K] @ Bt[N,K]^T + bias
// A row-major bf16, Bt transposed bf16 -> both tiles are contiguous 64B rows,
// staged purely with global_load_async_to_lds_b128 (4 per thread per k-tile).
// MODE 0: fp32 row-major out (+rowscale).  MODE 1: bf16 row-major out.
// MODE 2: bf16 out scattered to per-(b,h) transposed layout [(b*D+d)][S].
// ---------------------------------------------------------------------------
constexpr int BM = 128, BN = 128, BK = 32, LDT = 40;  // LDT rows = 80 B

template <int MODE, bool SCALED>
__global__ __launch_bounds__(256, 2)
void gemm_bf16_async(const __bf16* __restrict__ A, const __bf16* __restrict__ Bt,
                     const float* __restrict__ bias,
                     const float* __restrict__ rowscale,
                     void* __restrict__ Cout, int M, int N, int K) {
  __shared__ __bf16 As[2][BM * LDT];
  __shared__ __bf16 Bs[2][BN * LDT];

  const int tid  = threadIdx.x;
  const int lane = tid & 31;
  const int wv   = tid >> 5;
  const int wm   = (wv >> 1) << 5;
  const int wn   = (wv & 1) << 6;
  const int bm   = blockIdx.y * BM;
  const int bn   = blockIdx.x * BN;
  const int KT   = K / BK;  // 32 (power of two)

  // per-thread async lanes: rows tid>>2 and tid>>2+64, 16B chunk tid&3
  const int row  = tid >> 2;
  const int chk  = tid & 3;
  const char* gA0 = (const char*)(A + (size_t)(bm + row) * K) + chk * 16;
  const char* gA1 = (const char*)(A + (size_t)(bm + row + 64) * K) + chk * 16;
  const char* gB0 = (const char*)(Bt + (size_t)(bn + row) * K) + chk * 16;
  const char* gB1 = (const char*)(Bt + (size_t)(bn + row + 64) * K) + chk * 16;
  __bf16* dA0[2] = {&As[0][row * LDT + chk * 8], &As[1][row * LDT + chk * 8]};
  __bf16* dA1[2] = {&As[0][(row + 64) * LDT + chk * 8],
                    &As[1][(row + 64) * LDT + chk * 8]};
  __bf16* dB0[2] = {&Bs[0][row * LDT + chk * 8], &Bs[1][row * LDT + chk * 8]};
  __bf16* dB1[2] = {&Bs[0][(row + 64) * LDT + chk * 8],
                    &Bs[1][(row + 64) * LDT + chk * 8]};

#define ISSUE_KT(t)                                           \
  do {                                                        \
    const size_t off = (size_t)(t) * (BK * 2);                \
    async_copy_b128((void*)(gA0 + off), dA0[(t)&1]);          \
    async_copy_b128((void*)(gA1 + off), dA1[(t)&1]);          \
    async_copy_b128((void*)(gB0 + off), dB0[(t)&1]);          \
    async_copy_b128((void*)(gB1 + off), dB1[(t)&1]);          \
  } while (0)

  ISSUE_KT(0);
  ISSUE_KT(1);

  v8f acc[2][4] = {};

  for (int t = 0; t < KT; ++t) {
    wait_async_le4();   // tile t's 4 copies done (<=4 left = tile t+1)
    __syncthreads();
    const __bf16* as = As[t & 1];
    const __bf16* bs = Bs[t & 1];

    const int r = lane & 15;
    v16bf af[2], bf4[4];
#pragma unroll
    for (int tt = 0; tt < 2; ++tt)
      af[tt] = frag_a(&as[(wm + (tt << 4) + r) * LDT], lane);
#pragma unroll
    for (int tt = 0; tt < 4; ++tt)
      bf4[tt] = frag_b(&bs[(wn + (tt << 4) + r) * LDT], lane);
#pragma unroll
    for (int i = 0; i < 2; ++i)
#pragma unroll
      for (int j = 0; j < 4; ++j)
        acc[i][j] = wmma_bf16(af[i], bf4[j], acc[i][j]);

    __syncthreads();                 // buffer (t&1) free for overwrite
    ISSUE_KT((t + 2) & (KT - 1));    // branchless wrap (tail writes retired buf)
  }
#undef ISSUE_KT

  const int halfadd = (lane >> 4) << 3;
  const int cl      = lane & 15;

  float rsv[2][8];
#pragma unroll
  for (int i = 0; i < 2; ++i)
#pragma unroll
    for (int e = 0; e < 8; ++e)
      rsv[i][e] = SCALED ? rowscale[bm + wm + (i << 4) + halfadd + e] : 1.0f;

#pragma unroll
  for (int i = 0; i < 2; ++i)
#pragma unroll
    for (int j = 0; j < 4; ++j) {
      const int col  = bn + wn + (j << 4) + cl;
      const int row0 = bm + wm + (i << 4) + halfadd;
      if (MODE == 2) {
        // transposed bf16: dst[(b*D + col)][s], 8 consecutive s per lane
        __bf16 tmp[8];
#pragma unroll
        for (int e = 0; e < 8; ++e)
          tmp[e] = (__bf16)(acc[i][j][e] + bias[col]);
        __bf16* Cb = (__bf16*)Cout;
        size_t dst = ((size_t)((row0 >> 11) * D_ + col)) * S_ + (row0 & (S_ - 1));
        *(uint4*)(Cb + dst) = *(const uint4*)tmp;
      } else {
#pragma unroll
        for (int e = 0; e < 8; ++e) {
          int   rr = row0 + e;
          float v  = acc[i][j][e] * rsv[i][e] + bias[col];
          if (MODE == 0) ((float*)Cout)[(size_t)rr * N + col] = v;
          else           ((__bf16*)Cout)[(size_t)rr * N + col] = (__bf16)v;
        }
      }
    }
}

// ---------------------------------------------------------------------------
// Flash attention, one (b,h) per block column; q/k bf16 [B,S,D]; v bf16
// pre-transposed [(b*D+d)][S].  K/V slabs streamed with async-b128 copies,
// double buffered, branchless pipeline.  Softmax in the exp2 domain.
// Output o written bf16 [B,S,D].
// ---------------------------------------------------------------------------
constexpr int QS = 72;
constexpr int KS = 72;
constexpr int VS = 40;

__global__ __launch_bounds__(256, 1)
void moh_attention(const __bf16* __restrict__ q, const __bf16* __restrict__ k,
                   const __bf16* __restrict__ vt,
                   const float* __restrict__ temperature,
                   const float* __restrict__ qe, __bf16* __restrict__ o) {
  __shared__ __bf16 qs[128 * QS];
  __shared__ __bf16 kst[2][32 * KS];
  __shared__ __bf16 vtt[2][64 * VS];
  __shared__ __bf16 ps[8 * 16 * VS];

  const int tid  = threadIdx.x;
  const int lane = tid & 31;
  const int w    = tid >> 5;
  const int b    = blockIdx.y >> 4;
  const int h    = blockIdx.y & 15;
  const int q0   = blockIdx.x * 128;
  const size_t bhOff = (size_t)b * S_ * D_ + (size_t)h * HD_;

  const int krow = (w << 2) + (lane >> 3);
  const int kchk = lane & 7;
  const char* kg = (const char*)(k + bhOff + (size_t)krow * D_) + kchk * 16;
  __bf16* kd[2] = {&kst[0][krow * KS + kchk * 8], &kst[1][krow * KS + kchk * 8]};
  const int vrow = (w << 3) + (lane >> 2);
  const int vchk = lane & 3;
  const char* vg =
      (const char*)(vt + ((size_t)(b * D_ + h * HD_ + vrow)) * S_) + vchk * 16;
  __bf16* vd[2] = {&vtt[0][vrow * VS + vchk * 8], &vtt[1][vrow * VS + vchk * 8]};

  constexpr int NT = S_ / 32;

#define ISSUE_SLAB(t)                                                     \
  do {                                                                    \
    async_copy_b128((void*)(kg + (size_t)(t) * 32 * D_ * 2), kd[(t)&1]);  \
    async_copy_b128((void*)(vg + (size_t)(t) * 64), vd[(t)&1]);           \
  } while (0)

  ISSUE_SLAB(0);
  ISSUE_SLAB(1);

  // ---- stage scaled queries (wave-private rows): 2 threads per row ----
  {
    const float tv     = temperature[h];
    const float qscale = 0.125f * 1.44269504f * logf(1.0f + __expf(tv));
    const int   row    = tid >> 1;
    const int   half   = tid & 1;
    const __bf16* qp   = q + bhOff + (size_t)(q0 + row) * D_ + half * 32;
    union { uint4 u; __bf16 e[8]; } blk[4];
#pragma unroll
    for (int i = 0; i < 4; ++i) blk[i].u = *(const uint4*)(qp + i * 8);
    float vals[32];
    float ss = 0.0f;
#pragma unroll
    for (int i = 0; i < 32; ++i) {
      float f = (float)blk[i >> 3].e[i & 7];
      vals[i] = f;
      ss += f * f;
    }
    ss += __shfl_xor(ss, 1, 32);
    const float inv = rsqrtf(ss + 1e-12f);
    const float* qep = qe + h * HD_ + half * 32;
    __bf16* dst = &qs[row * QS + half * 32];
#pragma unroll
    for (int i = 0; i < 32; ++i)
      dst[i] = (__bf16)((vals[i] * inv + qep[i]) * qscale);
  }

  v16bf qa[2];
  {
    const int row = w * 16 + (lane & 15);
    qa[0] = frag_a(&qs[row * QS + 0], lane);
    qa[1] = frag_a(&qs[row * QS + 32], lane);
  }

  v8f   oacc[4] = {};
  float mrow[8], lrow[8];
#pragma unroll
  for (int i = 0; i < 8; ++i) { mrow[i] = -1e30f; lrow[i] = 0.0f; }

  for (int t = 0; t < NT; ++t) {
    wait_async_le2();
    __syncthreads();
    const __bf16* ksb = kst[t & 1];
    const __bf16* vsb = vtt[t & 1];

    v8f s0 = {}, s1 = {};
    {
      const int key = lane & 15;
      v16bf kb0 = frag_b(&ksb[key * KS + 0], lane);
      v16bf kb1 = frag_b(&ksb[key * KS + 32], lane);
      v16bf kb2 = frag_b(&ksb[(16 + key) * KS + 0], lane);
      v16bf kb3 = frag_b(&ksb[(16 + key) * KS + 32], lane);
      s0 = wmma_bf16(qa[0], kb0, s0);
      s0 = wmma_bf16(qa[1], kb1, s0);
      s1 = wmma_bf16(qa[0], kb2, s1);
      s1 = wmma_bf16(qa[1], kb3, s1);
    }

    v16bf vf[4];
    {
      const int row = lane & 15;
#pragma unroll
      for (int tt = 0; tt < 4; ++tt)
        vf[tt] = frag_b(&vsb[(tt * 16 + row) * VS], lane);
    }

    __bf16* pw = &ps[w * 16 * VS];
    float alpha[8];
#pragma unroll
    for (int i = 0; i < 8; ++i) {
      float a  = s0[i], c = s1[i];
      float mx = rmax16(fmaxf(a, c));
      float mn = fmaxf(mrow[i], mx);
      float al = exp2f(mrow[i] - mn);
      float p0 = exp2f(a - mn);
      float p1 = exp2f(c - mn);
      float rsm = rsum16(p0 + p1);
      lrow[i]  = lrow[i] * al + rsm;
      mrow[i]  = mn;
      alpha[i] = al;
      int r  = i + ((lane >> 4) << 3);
      int cc = lane & 15;
      pw[r * VS + cc]      = (__bf16)p0;
      pw[r * VS + 16 + cc] = (__bf16)p1;
    }
#pragma unroll
    for (int tt = 0; tt < 4; ++tt)
#pragma unroll
      for (int i = 0; i < 8; ++i) oacc[tt][i] *= alpha[i];

    {
      const int row = lane & 15;
      v16bf pf = frag_a(&pw[row * VS], lane);
#pragma unroll
      for (int tt = 0; tt < 4; ++tt)
        oacc[tt] = wmma_bf16(pf, vf[tt], oacc[tt]);
    }
    __syncthreads();
    ISSUE_SLAB((t + 2) & (NT - 1));
  }
#undef ISSUE_SLAB

  // ---- normalize and write o (bf16 [B,S,D]) ----
  __bf16* obp = o + bhOff;
#pragma unroll
  for (int i = 0; i < 8; ++i) {
    float invl = 1.0f / lrow[i];
    int   s    = q0 + w * 16 + i + ((lane >> 4) << 3);
    int   d0   = lane & 15;
#pragma unroll
    for (int tt = 0; tt < 4; ++tt)
      obp[(size_t)s * D_ + tt * 16 + d0] = (__bf16)(oacc[tt][i] * invl);
  }
}

// ---------------------------------------------------------------------------
// Per-token gate scale (one wave per token); o is bf16.
// ---------------------------------------------------------------------------
__global__ __launch_bounds__(256, 4)
void moh_gate_scale(const __bf16* __restrict__ o, const float* __restrict__ Wr,
                    const float* __restrict__ br, const float* __restrict__ Ws,
                    const float* __restrict__ bs, float* __restrict__ scale) {
  const int lane = threadIdx.x & 31;
  const int wid  = blockIdx.x * (blockDim.x >> 5) + (threadIdx.x >> 5);
  if (wid >= M_) return;
  const __bf16* orow = o + (size_t)wid * D_;

  float acc[17] = {};
  for (int d = lane; d < D_; d += 32) {
    float ov = (float)orow[d];
    const float* wr = Wr + (size_t)d * 15;
#pragma unroll
    for (int j = 0; j < 15; ++j) acc[j] += ov * wr[j];
    const float* ws = Ws + (size_t)d * 2;
    acc[15] += ov * ws[0];
    acc[16] += ov * ws[1];
  }
#pragma unroll
  for (int j = 0; j < 17; ++j)
#pragma unroll
    for (int m = 16; m >= 1; m >>= 1) acc[j] += __shfl_xor(acc[j], m, 32);

  if (lane == 0) {
    float g[15];
    float mx = -1e30f;
#pragma unroll
    for (int j = 0; j < 15; ++j) { g[j] = acc[j] + br[j]; mx = fmaxf(mx, g[j]); }
    float se = 0.0f;
#pragma unroll
    for (int j = 0; j < 15; ++j) { g[j] = __expf(g[j] - mx); se += g[j]; }
    float invse = 1.0f / se;
#pragma unroll
    for (int j = 0; j < 15; ++j) g[j] *= invse;

    float topsum = 0.0f;
    bool  used[15] = {};
#pragma unroll
    for (int r = 0; r < 3; ++r) {
      int   bi = 0;
      float bv = -1.0f;
#pragma unroll
      for (int j = 0; j < 15; ++j)
        if (!used[j] && g[j] > bv) { bv = g[j]; bi = j; }
      used[bi] = true;
      topsum += bv;
    }
    float s0 = acc[15] + bs[0], s1 = acc[16] + bs[1];
    float m2 = fmaxf(s0, s1);
    float e0 = __expf(s0 - m2), e1 = __expf(s1 - m2);
    float inv = 1.0f / (e0 + e1);
    float sw0 = e0 * inv, sw1 = e1 * inv;
    scale[wid] = 2.0f * sw0 * 1.0f + 2.0f * sw1 * 3.0f * topsum;  // SH=1, RH=3
  }
}

// ---------------------------------------------------------------------------
// Launcher
// ---------------------------------------------------------------------------
extern "C" void kernel_launch(void* const* d_in, const int* in_sizes, int n_in,
                              void* d_out, int out_size, void* d_ws,
                              size_t ws_size, hipStream_t stream) {
  (void)in_sizes; (void)n_in; (void)out_size; (void)ws_size;
  const float* x     = (const float*)d_in[0];
  const float* Wq    = (const float*)d_in[1];
  const float* bq    = (const float*)d_in[2];
  const float* Wk    = (const float*)d_in[3];
  const float* bk    = (const float*)d_in[4];
  const float* Wv    = (const float*)d_in[5];
  const float* bv    = (const float*)d_in[6];
  const float* Wproj = (const float*)d_in[7];
  const float* bproj = (const float*)d_in[8];
  const float* Wr    = (const float*)d_in[9];
  const float* br    = (const float*)d_in[10];
  const float* Wsg   = (const float*)d_in[11];
  const float* bsg   = (const float*)d_in[12];
  const float* temp  = (const float*)d_in[13];
  const float* qe    = (const float*)d_in[14];

  const size_t MD = (size_t)M_ * D_;   // 4M elements
  const size_t DD = (size_t)D_ * D_;   // 1M elements
  __bf16* xb  = (__bf16*)d_ws;         // 8 MB
  __bf16* Wqt = xb + MD;               // 2 MB each, transposed [N][K]
  __bf16* Wkt = Wqt + DD;
  __bf16* Wvt = Wkt + DD;
  __bf16* Wpt = Wvt + DD;
  __bf16* qb  = Wpt + DD;              // 8 MB
  __bf16* kb  = qb + MD;               // 8 MB
  __bf16* vtb = kb + MD;               // 8 MB (transposed layout)
  __bf16* ob  = vtb + MD;              // 8 MB
  float*  sc  = (float*)(ob + MD);     // 16 KB

  // one-time conversions
  cvt_f32_to_bf16<<<dim3((unsigned)(MD / 4 / 256)), 256, 0, stream>>>(x, xb);
  dim3 tgrid(D_ / 32, D_ / 32), tblk(256);
  transpose_to_bf16<<<tgrid, tblk, 0, stream>>>(Wq, Wqt);
  transpose_to_bf16<<<tgrid, tblk, 0, stream>>>(Wk, Wkt);
  transpose_to_bf16<<<tgrid, tblk, 0, stream>>>(Wv, Wvt);
  transpose_to_bf16<<<tgrid, tblk, 0, stream>>>(Wproj, Wpt);

  dim3 gblk(256), ggrid(D_ / BN, M_ / BM);
  gemm_bf16_async<1, false><<<ggrid, gblk, 0, stream>>>(xb, Wqt, bq, nullptr,
                                                        qb, M_, D_, D_);
  gemm_bf16_async<1, false><<<ggrid, gblk, 0, stream>>>(xb, Wkt, bk, nullptr,
                                                        kb, M_, D_, D_);
  gemm_bf16_async<2, false><<<ggrid, gblk, 0, stream>>>(xb, Wvt, bv, nullptr,
                                                        vtb, M_, D_, D_);

  moh_attention<<<dim3(S_ / 128, B_ * H_), 256, 0, stream>>>(qb, kb, vtb, temp,
                                                             qe, ob);

  moh_gate_scale<<<dim3(M_ / 8), 256, 0, stream>>>(ob, Wr, br, Wsg, bsg, sc);

  gemm_bf16_async<0, true><<<ggrid, gblk, 0, stream>>>(ob, Wpt, bproj, sc,
                                                       (float*)d_out, M_, D_, D_);
}
